// Attention_63556926046474
// MI455X (gfx1250) — matrix operand order repaired and checked by reading, not compiled, not run on hardware
//
#include <hip/hip_runtime.h>

typedef __attribute__((ext_vector_type(16))) __bf16 v16bf;
typedef __attribute__((ext_vector_type(8)))  float  v8f;

#define S_LEN 2048
#define D_DIM 2048
#define NH    16
#define HD    128
#define SCALE2 0.1275498454057111f   // (1/sqrt(128)) * log2(e)  -> exp2-domain softmax
#define NEG_BIG (-1.0e30f)
#define L2T 0.20762050593046f        // log2(10000)/64

union V16U { v16bf v; uint4 q[2]; unsigned short s[16]; };
union V8U  { v8f  v; float f[8]; };

__device__ __forceinline__ unsigned short f2bf(float x) {
  unsigned u = __float_as_uint(x);
  u += 0x7FFFu + ((u >> 16) & 1u);          // round-to-nearest-even
  return (unsigned short)(u >> 16);
}

#define WMMA_BF16(A, B, C) \
  __builtin_amdgcn_wmma_f32_16x16x32_bf16(false, (A), false, (B), (short)0, (C), false, false)

// ============ preprocess 1: RoPE(Q,K) -> bf16, layout [b,h,s,128] ============
// One sincos per (s, i) serves both Q and K, computed ONCE per token
// (vs. once per q-tile-block = 16x redundant if fused into the FA loop).
__global__ __launch_bounds__(256)
void rope_qk_bf16(const float* __restrict__ xq, const float* __restrict__ xk,
                  unsigned short* __restrict__ Qr, unsigned short* __restrict__ Kr)
{
  const int h = blockIdx.y, b = blockIdx.z;
  const int t = (int)threadIdx.x;
  const int i = t & 63;                       // freq index < 64
  const int s = blockIdx.x * 4 + (t >> 6);    // 4 rows per block
  const size_t iof = ((size_t)b * S_LEN + s) * D_DIM + (size_t)h * HD;
  const size_t oof = (((size_t)(b * NH) + h) * S_LEN + s) * HD;
  const float invf = exp2f(-(float)i * L2T);
  float sn, cs;
  __sincosf((float)s * invf, &sn, &cs);
  {
    const float x1 = xq[iof + i], x2 = xq[iof + i + 64];
    Qr[oof + i]      = f2bf(x1 * cs + x2 * sn);
    Qr[oof + i + 64] = f2bf(x2 * cs - x1 * sn);
  }
  {
    const float x1 = xk[iof + i], x2 = xk[iof + i + 64];
    Kr[oof + i]      = f2bf(x1 * cs + x2 * sn);
    Kr[oof + i + 64] = f2bf(x2 * cs - x1 * sn);
  }
}

// ============ preprocess 2: V -> bf16 transposed, layout [b,h,a,s] ============
__global__ __launch_bounds__(256)
void v_transpose_bf16(const float* __restrict__ xv, unsigned short* __restrict__ Vt)
{
  const int h = blockIdx.y, b = blockIdx.z;
  const int t = (int)threadIdx.x;
  const int a  = t & 127;
  const int s0 = blockIdx.x * 64 + (t >> 7) * 32;
  const float* src = xv + ((size_t)b * S_LEN) * D_DIM + (size_t)h * HD + a;
  unsigned short* dst = Vt + (((size_t)(b * NH) + h) * HD + a) * S_LEN + s0;
  #pragma unroll
  for (int g = 0; g < 4; ++g) {
    union { uint4 q; unsigned short us[8]; } pk;
    #pragma unroll
    for (int j = 0; j < 8; ++j)
      pk.us[j] = f2bf(src[(size_t)(s0 + g * 8 + j) * D_DIM]);
    *(uint4*)(dst + g * 8) = pk.q;
  }
}

// ============ FA2 core: one 32-key block ============
// Transposed GEMMs (S^T = K*Q^T, O^T = V^T*P^T) so each lane owns one query
// column; softmax state is a per-lane scalar. Operands load global->VGPR as
// 16B b128 in exact WMMA layout; 32-bit offsets keep address math off VALU.
template <bool MASK>
__device__ __forceinline__ void fa_iter(const int kbase, const int qrow,
                                        const int lq, const int hl,
                                        const unsigned short* __restrict__ Kb,
                                        const unsigned short* __restrict__ Vb,
                                        const V16U* __restrict__ qop,
                                        v8f* __restrict__ acc,
                                        float& mrun, float& lrun)
{
  const int cofs = hl * 8;
  // ---- S^T = K x Q^T : two 16x16 key tiles, K-dim = 4 x 32 channels
  v8f sc[2];
  {
    const int k0 = (kbase + lq) * HD + cofs;          // 32-bit offsets
    #pragma unroll
    for (int t = 0; t < 2; ++t) {
      v8f c = {0.f, 0.f, 0.f, 0.f, 0.f, 0.f, 0.f, 0.f};
      const int kt = k0 + t * (16 * HD);
      #pragma unroll
      for (int ch = 0; ch < 4; ++ch) {
        V16U aop;
        aop.q[0] = *(const uint4*)(Kb + (kt + ch * 32));
        aop.q[1] = *(const uint4*)(Kb + (kt + ch * 32 + 16));
        c = WMMA_BF16(aop.v, qop[ch].v, c);
      }
      sc[t] = c;
    }
  }
  // ---- online softmax in exp2 domain (per-lane scalar state)
  float sv[16];
  float mloc = NEG_BIG;
  #pragma unroll
  for (int t = 0; t < 2; ++t) {
    V8U u; u.v = sc[t];
    #pragma unroll
    for (int r = 0; r < 8; ++r) {
      float s = u.f[r] * SCALE2;
      if (MASK) {
        const int kg = kbase + t * 16 + hl * 8 + r;
        s = (kg <= qrow) ? s : NEG_BIG;
      }
      sv[t * 8 + r] = s;
      mloc = fmaxf(mloc, s);
    }
  }
  const float mpair = fmaxf(mloc, __shfl_xor(mloc, 16, 32)); // lanes l, l^16 share q
  const float mnew  = fmaxf(mrun, mpair);
  float pvv[16];
  float psum = 0.f;
  #pragma unroll
  for (int i = 0; i < 16; ++i) { const float p = exp2f(sv[i] - mnew); pvv[i] = p; psum += p; }
  psum += __shfl_xor(psum, 16, 32);
  // Rescale only when some lane's running max moved (wave-uniform branch).
  // If mpair <= mrun on ALL lanes, mnew == mrun exactly => resc == 1.0.
  if (__ballot(mpair > mrun) != 0ull) {
    const float resc = exp2f(mrun - mnew);
    lrun = lrun * resc + psum;
    #pragma unroll
    for (int ab = 0; ab < 8; ++ab) {
      V8U u; u.v = acc[ab];
      #pragma unroll
      for (int r = 0; r < 8; ++r) u.f[r] *= resc;
      acc[ab] = u.v;
    }
  } else {
    lrun += psum;
  }
  mrun = mnew;
  // ---- P^T B-operand from C-layout scores: one half-swap shuffle per reg
  V16U pop;
  #pragma unroll
  for (int r = 0; r < 8; ++r) {
    const float a0v = pvv[r], a1v = pvv[8 + r];
    const float b0v = __shfl_xor(a0v, 16, 32);
    const float b1v = __shfl_xor(a1v, 16, 32);
    pop.v[r]     = (__bf16)(hl ? b1v : a0v);
    pop.v[8 + r] = (__bf16)(hl ? a1v : b0v);
  }
  // ---- O^T += V^T x P^T (8 channel blocks), V^T rows contiguous in global
  int voff = lq * S_LEN + kbase + hl * 16;            // 32-bit offset walk
  #pragma unroll
  for (int ab = 0; ab < 8; ++ab) {
    V16U vop;
    vop.q[0] = *(const uint4*)(Vb + voff);
    vop.q[1] = *(const uint4*)(Vb + (voff + 8));
    acc[ab] = WMMA_BF16(vop.v, pop.v, acc[ab]);
    voff += 16 * S_LEN;
  }
}

__global__ __launch_bounds__(256)
void fa2_core(const unsigned short* __restrict__ Qr,
              const unsigned short* __restrict__ Kr,
              const unsigned short* __restrict__ Vt,
              float* __restrict__ out)
{
  const int qt  = blockIdx.x;        // 128 query rows per block (8 waves x 16)
  const int h   = blockIdx.y;
  const int b   = blockIdx.z;
  const int tid = (int)threadIdx.x;
  const int lane = tid & 31;
  const int wv   = tid >> 5;
  const int lq   = lane & 15;
  const int hl   = lane >> 4;

  const int q0w  = qt * 128 + wv * 16;
  const int qrow = q0w + lq;

  const int bhS = (b * NH + h) * S_LEN;
  const unsigned short* Kb = Kr + (size_t)bhS * HD;                 // SGPR bases
  const unsigned short* Vb = Vt + (size_t)(b * NH + h) * HD * S_LEN;

  // Q -> B-operand registers (4 chunks), already RoPE'd bf16
  V16U qop[4];
  {
    const unsigned short* Qb = Qr + (size_t)bhS * HD;
    const int qoff = qrow * HD + hl * 8;
    #pragma unroll
    for (int ch = 0; ch < 4; ++ch) {
      qop[ch].q[0] = *(const uint4*)(Qb + (qoff + ch * 32));
      qop[ch].q[1] = *(const uint4*)(Qb + (qoff + ch * 32 + 16));
    }
  }

  v8f acc[8];
  {
    V8U z;
    #pragma unroll
    for (int r = 0; r < 8; ++r) z.f[r] = 0.f;
    #pragma unroll
    for (int i = 0; i < 8; ++i) acc[i] = z.v;
  }
  float mrun = NEG_BIG, lrun = 0.f;

  // exact per-wave causal bound: nfull unmasked tiles + exactly one masked tile
  const int nfull = (q0w + 1) >> 5;
  for (int kb = 0; kb < nfull; ++kb) {
    const int kbase = kb * 32;
    __builtin_prefetch(Kb + ((kbase + 32 + lq) * HD), 0, 3);
    __builtin_prefetch(Vb + (lq * S_LEN + kbase + 32), 0, 3);
    fa_iter<false>(kbase, qrow, lq, hl, Kb, Vb, qop, acc, mrun, lrun);
  }
  fa_iter<true>(nfull * 32, qrow, lq, hl, Kb, Vb, qop, acc, mrun, lrun);

  // normalize, store: out[b][qrow][h*128 + a]
  const float invl = 1.0f / lrun;
  float* op = out + ((size_t)b * S_LEN + qrow) * D_DIM + (size_t)h * HD;
  #pragma unroll
  for (int ab = 0; ab < 8; ++ab) {
    V8U u; u.v = acc[ab];
    const int a = ab * 16 + hl * 8;
    *(float4*)(op + a)     = make_float4(u.f[0] * invl, u.f[1] * invl, u.f[2] * invl, u.f[3] * invl);
    *(float4*)(op + a + 4) = make_float4(u.f[4] * invl, u.f[5] * invl, u.f[6] * invl, u.f[7] * invl);
  }
}

extern "C" void kernel_launch(void* const* d_in, const int* in_sizes, int n_in,
                              void* d_out, int out_size, void* d_ws, size_t ws_size,
                              hipStream_t stream) {
  const float* xq = (const float*)d_in[0];
  const float* xk = (const float*)d_in[1];
  const float* xv = (const float*)d_in[2];
  float* o = (float*)d_out;
  const int B = in_sizes[0] / (S_LEN * D_DIM);
  const size_t NTOK = (size_t)B * NH * S_LEN * HD;     // bf16 elems per tensor
  unsigned short* Qr = (unsigned short*)d_ws;          // 3 * NTOK * 2 bytes of ws used
  unsigned short* Kr = Qr + NTOK;
  unsigned short* Vt = Kr + NTOK;

  rope_qk_bf16    <<<dim3(S_LEN / 4,   NH, B), 256, 0, stream>>>(xq, xk, Qr, Kr);
  v_transpose_bf16<<<dim3(S_LEN / 64,  NH, B), 256, 0, stream>>>(xv, Vt);
  fa2_core        <<<dim3(S_LEN / 128, NH, B), 256, 0, stream>>>(Qr, Kr, Vt, o);

  (void)n_in; (void)out_size; (void)ws_size;
}
